// NNBase_19636590477825
// MI455X (gfx1250) — matrix operand order repaired and checked by reading, not compiled, not run on hardware
//
#include <hip/hip_runtime.h>
#include <hip/hip_bf16.h>

#define T_DIM 128
#define N_DIM 64
#define H_DIM 1024
#define A_DIM 16
#define G3    3072

typedef __attribute__((ext_vector_type(16))) __bf16 v16bf;
typedef __attribute__((ext_vector_type(8)))  float  v8f;

// D = A(16x32 bf16) x B(32x16 bf16) + C(16x16 f32), CDNA5 wave32 WMMA.
__device__ __forceinline__ v8f wmma_bf16(v16bf a, v16bf b, v8f c) {
  return __builtin_amdgcn_wmma_f32_16x16x32_bf16(false, a, false, b, (short)0, c,
                                                 false, false);
}

__device__ __forceinline__ __bf16 to_bf(float x) { return (__bf16)x; }

__device__ __forceinline__ float sigm(float x) { return 1.0f / (1.0f + __expf(-x)); }
__device__ __forceinline__ float tanh_fast(float x) {
  return 2.0f / (1.0f + __expf(-2.0f * x)) - 1.0f;
}

// Load one A operand (16x32 tile, ISA 16-bit A layout) from an f32 row-major
// matrix, scaling by a per-lane (per-row) factor, converting to bf16.
// A layout: lane -> M = lane&15 ; element e -> K = (lane>>4)*8 + (e&7) + (e>=8)*16
__device__ __forceinline__ v16bf load_a_f32(const float* __restrict__ M, int row0,
                                            int col0, int stride, int lane,
                                            float scale) {
  int r = row0 + (lane & 15);
  const float* p = M + (size_t)r * stride + col0 + ((lane >> 4) * 8);
  v16bf a;
#pragma unroll
  for (int e = 0; e < 8; ++e) a[e] = to_bf(p[e] * scale);
#pragma unroll
  for (int e = 0; e < 8; ++e) a[8 + e] = to_bf(p[16 + e] * scale);
  return a;
}

// Scatter one scalar into a packed A-layout bf16 tile buffer (tiles of 512
// elements, tile index = Mblk*kblks + kblk).  Inverse of the A-layout map.
__device__ __forceinline__ void store_a_packed(__bf16* __restrict__ dst, int kblks,
                                               int M, int K, float val) {
  int kblk = K >> 5;
  int Kl = K & 31;
  int lane_a = (M & 15) + (((Kl >> 3) & 1) << 4);
  int e = (Kl & 7) + ((Kl >> 4) << 3);
  dst[((size_t)((M >> 4) * kblks + kblk)) * 512 + lane_a * 16 + e] = to_bf(val);
}

// ---------------- packing kernels (run once, fully parallel) ----------------

__global__ __launch_bounds__(256) void k_pack_A(const float* __restrict__ src,
                                                __bf16* __restrict__ dst, int mblks,
                                                int kblks, int stride) {
  int tid = blockIdx.x * blockDim.x + threadIdx.x;
  int lane = tid & 31;
  int idx = tid >> 5;
  if (idx >= mblks * kblks) return;
  int kblk = idx % kblks;
  int mblk = idx / kblks;
  v16bf a = load_a_f32(src, mblk * 16, kblk * 32, stride, lane, 1.0f);
  *(v16bf*)(dst + (size_t)idx * 512 + lane * 16) = a;
}

// Pack W (R x Kreal row-major) as B-operand tiles of W^T:
// N = nblk*16 + (lane&15) (row of W), K = kblk*32 + (lane>>4)*8 + (e&7) + (e>>3)*16.
__global__ __launch_bounds__(256) void k_pack_B(const float* __restrict__ W,
                                                __bf16* __restrict__ dst, int nblks,
                                                int kblks, int Kreal) {
  int tid = blockIdx.x * blockDim.x + threadIdx.x;
  int lane = tid & 31;
  int idx = tid >> 5;
  if (idx >= nblks * kblks) return;
  int kblk = idx % kblks;
  int nblk = idx / kblks;
  int row = nblk * 16 + (lane & 15);
  int c0 = kblk * 32 + ((lane >> 4) * 8);
  const float* p = W + (size_t)row * Kreal + c0;
  v16bf b;
#pragma unroll
  for (int e = 0; e < 16; ++e) {
    int ko = (e & 7) + ((e >> 3) << 4);
    float v = (c0 + ko < Kreal) ? p[ko] : 0.0f;
    b[e] = to_bf(v);
  }
  *(v16bf*)(dst + (size_t)idx * 512 + lane * 16) = b;
}

// ---------------- gi precompute: gi = c @ w_ih^T + b_ih  (8192 x 3072) -----
// One wave -> 16x64 output strip (4 n-tiles, shared A), K=1024.
// Ping-pong double buffering (unroll-2): zero register rotation copies.
__global__ __launch_bounds__(256) void k_gi_gemm(const __bf16* __restrict__ cA,
                                                 const __bf16* __restrict__ wB,
                                                 const float* __restrict__ bias,
                                                 float* __restrict__ gi) {
  int lane = threadIdx.x & 31;
  int task = blockIdx.x * 8 + (threadIdx.x >> 5);  // 512 mblk * 48 ngrp
  if (task >= 512 * 48) return;
  int mblk = task / 48;
  int nb0 = (task % 48) * 4;
  const __bf16* ca = cA + (size_t)mblk * 32 * 512 + lane * 16;
  const __bf16* wb = wB + (size_t)nb0 * 32 * 512 + lane * 16;

#define GI_LDA(kc) (*(const v16bf*)(ca + (size_t)(kc) * 512))
#define GI_LDB(j, kc) (*(const v16bf*)(wb + (size_t)((j) * 32 + (kc)) * 512))

  v8f c0 = {}, c1 = {}, c2 = {}, c3 = {};
  v16bf a0 = GI_LDA(0);
  v16bf p0 = GI_LDB(0, 0), p1 = GI_LDB(1, 0), p2 = GI_LDB(2, 0), p3 = GI_LDB(3, 0);
  for (int kc = 0; kc < 30; kc += 2) {
    v16bf a1 = GI_LDA(kc + 1);
    v16bf q0 = GI_LDB(0, kc + 1), q1 = GI_LDB(1, kc + 1);
    v16bf q2 = GI_LDB(2, kc + 1), q3 = GI_LDB(3, kc + 1);
    c0 = wmma_bf16(a0, p0, c0); c1 = wmma_bf16(a0, p1, c1);
    c2 = wmma_bf16(a0, p2, c2); c3 = wmma_bf16(a0, p3, c3);
    a0 = GI_LDA(kc + 2);
    p0 = GI_LDB(0, kc + 2); p1 = GI_LDB(1, kc + 2);
    p2 = GI_LDB(2, kc + 2); p3 = GI_LDB(3, kc + 2);
    c0 = wmma_bf16(a1, q0, c0); c1 = wmma_bf16(a1, q1, c1);
    c2 = wmma_bf16(a1, q2, c2); c3 = wmma_bf16(a1, q3, c3);
  }
  {  // set0 holds chunk 30; load chunk 31 into set1 and finish
    v16bf a1 = GI_LDA(31);
    v16bf q0 = GI_LDB(0, 31), q1 = GI_LDB(1, 31);
    v16bf q2 = GI_LDB(2, 31), q3 = GI_LDB(3, 31);
    c0 = wmma_bf16(a0, p0, c0); c1 = wmma_bf16(a0, p1, c1);
    c2 = wmma_bf16(a0, p2, c2); c3 = wmma_bf16(a0, p3, c3);
    c0 = wmma_bf16(a1, q0, c0); c1 = wmma_bf16(a1, q1, c1);
    c2 = wmma_bf16(a1, q2, c2); c3 = wmma_bf16(a1, q3, c3);
  }
#undef GI_LDA
#undef GI_LDB

  int rb = mblk * 16 + 8 * (lane >> 4);
  v8f accs[4] = {c0, c1, c2, c3};
#pragma unroll
  for (int j = 0; j < 4; ++j) {
    int col = (nb0 + j) * 16 + (lane & 15);
    float bv = bias[col];
#pragma unroll
    for (int v = 0; v < 8; ++v) gi[(size_t)(rb + v) * G3 + col] = accs[j][v] + bv;
  }
}

// ---------------- state init / final-state epilogue ------------------------
__global__ __launch_bounds__(256) void k_init(const float* __restrict__ hxs,
                                              float* __restrict__ h0,
                                              float* __restrict__ pol0) {
  int i = blockIdx.x * blockDim.x + threadIdx.x;
  if (i >= N_DIM * H_DIM) return;
  int n = i / H_DIM, j = i % H_DIM;
  h0[i] = hxs[(size_t)n * 2 * H_DIM + j];
  pol0[i] = hxs[(size_t)n * 2 * H_DIM + H_DIM + j];
}

__global__ __launch_bounds__(256) void k_epilogue(const float* __restrict__ hf,
                                                  const float* __restrict__ pf,
                                                  float* __restrict__ out) {
  int i = blockIdx.x * blockDim.x + threadIdx.x;
  if (i >= N_DIM * H_DIM) return;
  int n = i / H_DIM, j = i % H_DIM;
  out[(size_t)n * 2 * H_DIM + j] = hf[i];
  out[(size_t)n * 2 * H_DIM + H_DIM + j] = pf[i];
}

// ---------------- per-timestep GRU kernel (serial critical path) -----------
// 64 blocks x 8 waves.  which = bid>>5, mblk = (bid>>3)&3, wave -> one column
// block nb.  Each wave computes the three gate tiles (r/z/n) for its 16
// columns over K=1024 (ping-pong pipelined), applies the GRU nonlinearity,
// writes h_new.  The h-GRU additionally scatters h_prev*mask into the packed
// bf16 A-layout history buffer consumed by the post-scan policy MLP.
__global__ __launch_bounds__(256) void k_step(
    const float* __restrict__ gih, const float* __restrict__ gip,
    const __bf16* __restrict__ wBhh_h, const __bf16* __restrict__ wBhh_p,
    const float* __restrict__ bhh_h, const float* __restrict__ bhh_p,
    const float* __restrict__ masks, const float* __restrict__ h_in,
    float* __restrict__ h_out, const float* __restrict__ pol_in,
    float* __restrict__ pol_out, __bf16* __restrict__ h_histP,
    float* __restrict__ out_policy, int t) {
  int lane = threadIdx.x & 31;
  int wv = threadIdx.x >> 5;
  int bid = blockIdx.x;

  int which = bid >> 5;              // 0 = h-GRU, 1 = pol-GRU
  int rem = bid & 31;
  int mblk = rem >> 3;               // 0..3 (batch tile)
  int nb = (rem & 7) * 8 + wv;       // 0..63 (column block of H)
  int m0 = mblk * 16;
  const float* gi = which ? gip : gih;
  const __bf16* wB = which ? wBhh_p : wBhh_h;
  const float* bhh = which ? bhh_p : bhh_h;
  const float* sin_ = which ? pol_in : h_in;
  float* sout = which ? pol_out : h_out;

  int ar = m0 + (lane & 15);
  float msk_a = masks[t * N_DIM + ar];
  const __bf16* wbb = wB + (size_t)nb * 32 * 512 + lane * 16;

#define ST_LDA(kc) load_a_f32(sin_, m0, (kc) * 32, H_DIM, lane, msk_a)
#define ST_LDB(g, kc) \
  (*(const v16bf*)(wbb + (size_t)((g) * 64 * 32 + (kc)) * 512))

  v8f g0 = {}, g1 = {}, g2 = {};
  v16bf a0 = ST_LDA(0);
  v16bf p0 = ST_LDB(0, 0), p1 = ST_LDB(1, 0), p2 = ST_LDB(2, 0);
  for (int kc = 0; kc < 30; kc += 2) {
    v16bf a1 = ST_LDA(kc + 1);
    v16bf q0 = ST_LDB(0, kc + 1), q1 = ST_LDB(1, kc + 1), q2 = ST_LDB(2, kc + 1);
    g0 = wmma_bf16(a0, p0, g0);
    g1 = wmma_bf16(a0, p1, g1);
    g2 = wmma_bf16(a0, p2, g2);
    a0 = ST_LDA(kc + 2);
    p0 = ST_LDB(0, kc + 2); p1 = ST_LDB(1, kc + 2); p2 = ST_LDB(2, kc + 2);
    g0 = wmma_bf16(a1, q0, g0);
    g1 = wmma_bf16(a1, q1, g1);
    g2 = wmma_bf16(a1, q2, g2);
  }
  {
    v16bf a1 = ST_LDA(31);
    v16bf q0 = ST_LDB(0, 31), q1 = ST_LDB(1, 31), q2 = ST_LDB(2, 31);
    g0 = wmma_bf16(a0, p0, g0);
    g1 = wmma_bf16(a0, p1, g1);
    g2 = wmma_bf16(a0, p2, g2);
    g0 = wmma_bf16(a1, q0, g0);
    g1 = wmma_bf16(a1, q1, g1);
    g2 = wmma_bf16(a1, q2, g2);
  }
#undef ST_LDA
#undef ST_LDB

  int col = nb * 16 + (lane & 15);
  float br = bhh[col], bz = bhh[col + H_DIM], bn = bhh[col + 2 * H_DIM];
  int rb = m0 + 8 * (lane >> 4);
#pragma unroll
  for (int v = 0; v < 8; ++v) {
    int r = rb + v;
    float mv = masks[t * N_DIM + r];
    const float* gr = gi + (size_t)(t * N_DIM + r) * G3;
    float hprev = sin_[(size_t)r * H_DIM + col] * mv;
    float rr = sigm(gr[col] + g0[v] + br);
    float zz = sigm(gr[col + H_DIM] + g1[v] + bz);
    float nn = tanh_fast(gr[col + 2 * H_DIM] + rr * (g2[v] + bn));
    float hv = (1.0f - zz) * nn + zz * hprev;
    sout[(size_t)r * H_DIM + col] = hv;
    if (which) {
      out_policy[(size_t)(t * N_DIM + r) * H_DIM + col] = hv;
    } else {
      // masked input state h_{t-1}*m_t == the MLP's A operand for row t*64+r
      store_a_packed(h_histP, 32, t * N_DIM + r, col, hprev);
    }
  }
}

// ---------------- post-scan policy MLP (fully parallel over T*N) -----------
// hidden = tanh([h_hist, a]*m @ p_w1^T + b1)  (8192 x 1024, K=1040)
// One wave -> 16x64 strip; h-part streams packed bf16 A tiles (pre-masked).
__global__ __launch_bounds__(256) void k_mlp1(const __bf16* __restrict__ h_histP,
                                              const __bf16* __restrict__ pw1B,
                                              const float* __restrict__ p_b1,
                                              const float* __restrict__ masks,
                                              const float* __restrict__ prev_a,
                                              __bf16* __restrict__ hiddenP) {
  int lane = threadIdx.x & 31;
  int task = blockIdx.x * 8 + (threadIdx.x >> 5);  // 512 mblk * 16 ngrp
  if (task >= 512 * 16) return;
  int mblk = task / 16;
  int nb0 = (task % 16) * 4;
  const __bf16* ha = h_histP + (size_t)mblk * 32 * 512 + lane * 16;
  const __bf16* wb = pw1B + (size_t)nb0 * 33 * 512 + lane * 16;

#define M1_LDA(kc) (*(const v16bf*)(ha + (size_t)(kc) * 512))
#define M1_LDB(j, kc) (*(const v16bf*)(wb + (size_t)((j) * 33 + (kc)) * 512))

  v8f c0 = {}, c1 = {}, c2 = {}, c3 = {};
  v16bf a0 = M1_LDA(0);
  v16bf p0 = M1_LDB(0, 0), p1 = M1_LDB(1, 0), p2 = M1_LDB(2, 0), p3 = M1_LDB(3, 0);
  for (int kc = 0; kc < 30; kc += 2) {
    v16bf a1 = M1_LDA(kc + 1);
    v16bf q0 = M1_LDB(0, kc + 1), q1 = M1_LDB(1, kc + 1);
    v16bf q2 = M1_LDB(2, kc + 1), q3 = M1_LDB(3, kc + 1);
    c0 = wmma_bf16(a0, p0, c0); c1 = wmma_bf16(a0, p1, c1);
    c2 = wmma_bf16(a0, p2, c2); c3 = wmma_bf16(a0, p3, c3);
    a0 = M1_LDA(kc + 2);
    p0 = M1_LDB(0, kc + 2); p1 = M1_LDB(1, kc + 2);
    p2 = M1_LDB(2, kc + 2); p3 = M1_LDB(3, kc + 2);
    c0 = wmma_bf16(a1, q0, c0); c1 = wmma_bf16(a1, q1, c1);
    c2 = wmma_bf16(a1, q2, c2); c3 = wmma_bf16(a1, q3, c3);
  }
  {
    v16bf a1 = M1_LDA(31);
    v16bf q0 = M1_LDB(0, 31), q1 = M1_LDB(1, 31);
    v16bf q2 = M1_LDB(2, 31), q3 = M1_LDB(3, 31);
    c0 = wmma_bf16(a0, p0, c0); c1 = wmma_bf16(a0, p1, c1);
    c2 = wmma_bf16(a0, p2, c2); c3 = wmma_bf16(a0, p3, c3);
    c0 = wmma_bf16(a1, q0, c0); c1 = wmma_bf16(a1, q1, c1);
    c2 = wmma_bf16(a1, q2, c2); c3 = wmma_bf16(a1, q3, c3);
  }
  {  // K chunk 32: columns 1024..1039 = action one-hot * mask, rest zero
    int rg = mblk * 16 + (lane & 15);
    float mv = masks[rg];
    v16bf a2;
    int cb = 1024 + (lane >> 4) * 8;
#pragma unroll
    for (int e = 0; e < 16; ++e) {
      int k = cb + (e & 7) + ((e >> 3) * 16);
      float val = (k < H_DIM + A_DIM)
                      ? prev_a[(size_t)rg * A_DIM + (k - H_DIM)] * mv
                      : 0.0f;
      a2[e] = to_bf(val);
    }
    c0 = wmma_bf16(a2, M1_LDB(0, 32), c0);
    c1 = wmma_bf16(a2, M1_LDB(1, 32), c1);
    c2 = wmma_bf16(a2, M1_LDB(2, 32), c2);
    c3 = wmma_bf16(a2, M1_LDB(3, 32), c3);
  }
#undef M1_LDA
#undef M1_LDB

  int rb = mblk * 16 + 8 * (lane >> 4);
  v8f accs[4] = {c0, c1, c2, c3};
#pragma unroll
  for (int j = 0; j < 4; ++j) {
    int col = (nb0 + j) * 16 + (lane & 15);
    float bv = p_b1[col];
#pragma unroll
    for (int v = 0; v < 8; ++v)
      store_a_packed(hiddenP, 32, rb + v, col, tanh_fast(accs[j][v] + bv));
  }
}

// p_dist = hidden @ p_w2^T + b2   (8192 x 16, K=1024); one wave per 16 rows.
// Two accumulators so back-to-back WMMAs are independent.
__global__ __launch_bounds__(256) void k_mlp2(const __bf16* __restrict__ hiddenP,
                                              const __bf16* __restrict__ pw2B,
                                              const float* __restrict__ p_b2,
                                              float* __restrict__ out_pdist) {
  int lane = threadIdx.x & 31;
  int mblk = blockIdx.x * 8 + (threadIdx.x >> 5);  // 512 row blocks
  if (mblk >= 512) return;
  const __bf16* ha = hiddenP + (size_t)mblk * 32 * 512 + lane * 16;
  const __bf16* wb = pw2B + lane * 16;

#define M2_LDA(kc) (*(const v16bf*)(ha + (size_t)(kc) * 512))
#define M2_LDB(kc) (*(const v16bf*)(wb + (size_t)(kc) * 512))

  v8f accA = {}, accB = {};
  v16bf a0 = M2_LDA(0), b0 = M2_LDB(0);
  for (int kc = 0; kc < 30; kc += 2) {
    v16bf a1 = M2_LDA(kc + 1), b1 = M2_LDB(kc + 1);
    accA = wmma_bf16(a0, b0, accA);
    a0 = M2_LDA(kc + 2); b0 = M2_LDB(kc + 2);
    accB = wmma_bf16(a1, b1, accB);
  }
  {
    v16bf a1 = M2_LDA(31), b1 = M2_LDB(31);
    accA = wmma_bf16(a0, b0, accA);
    accB = wmma_bf16(a1, b1, accB);
  }
#undef M2_LDA
#undef M2_LDB

  int col = lane & 15;
  float bv = p_b2[col];
  int rb = mblk * 16 + 8 * (lane >> 4);
#pragma unroll
  for (int v = 0; v < 8; ++v)
    out_pdist[(size_t)(rb + v) * A_DIM + col] = accA[v] + accB[v] + bv;
}

// ---------------------------------------------------------------------------
extern "C" void kernel_launch(void* const* d_in, const int* in_sizes, int n_in,
                              void* d_out, int out_size, void* d_ws, size_t ws_size,
                              hipStream_t stream) {
  (void)in_sizes; (void)n_in; (void)out_size; (void)ws_size;
  const float* c      = (const float*)d_in[0];
  const float* hxs    = (const float*)d_in[1];
  const float* masks  = (const float*)d_in[2];
  const float* prev_a = (const float*)d_in[3];
  const float* w_ih_h = (const float*)d_in[4];
  const float* w_hh_h = (const float*)d_in[5];
  const float* b_ih_h = (const float*)d_in[6];
  const float* b_hh_h = (const float*)d_in[7];
  const float* w_ih_p = (const float*)d_in[8];
  const float* w_hh_p = (const float*)d_in[9];
  const float* b_ih_p = (const float*)d_in[10];
  const float* b_hh_p = (const float*)d_in[11];
  const float* p_w1   = (const float*)d_in[12];
  const float* p_b1   = (const float*)d_in[13];
  const float* p_w2   = (const float*)d_in[14];
  const float* p_b2   = (const float*)d_in[15];

  char* ws = (char*)d_ws;
  size_t off = 0;
  auto alloc = [&](size_t bytes) -> void* {
    void* p = ws + off;
    off = (off + bytes + 255) & ~(size_t)255;
    return p;
  };
  __bf16* cA     = (__bf16*)alloc((size_t)512 * 32 * 512 * 2);   // packed c (A)
  __bf16* wBihh  = (__bf16*)alloc((size_t)192 * 32 * 512 * 2);   // w_ih_h^T (B)
  __bf16* wBhhh  = (__bf16*)alloc((size_t)192 * 32 * 512 * 2);   // w_hh_h^T (B)
  __bf16* wBihp  = (__bf16*)alloc((size_t)192 * 32 * 512 * 2);   // w_ih_p^T (B)
  __bf16* wBhhp  = (__bf16*)alloc((size_t)192 * 32 * 512 * 2);   // w_hh_p^T (B)
  __bf16* pw1B   = (__bf16*)alloc((size_t)64 * 33 * 512 * 2);    // p_w1^T   (B)
  __bf16* pw2B   = (__bf16*)alloc((size_t)1 * 32 * 512 * 2);     // p_w2^T   (B)
  float*  gih    = (float*)alloc((size_t)8192 * G3 * 4);         // c@w_ih_h^T+b
  float*  gip    = (float*)alloc((size_t)8192 * G3 * 4);         // c@w_ih_p^T+b
  float*  hbuf   = (float*)alloc((size_t)2 * N_DIM * H_DIM * 4); // double-buffer
  float*  polbuf = (float*)alloc((size_t)2 * N_DIM * H_DIM * 4);
  __bf16* h_histP= (__bf16*)alloc((size_t)512 * 32 * 512 * 2);   // masked h hist
  __bf16* hiddenP= (__bf16*)alloc((size_t)512 * 32 * 512 * 2);   // MLP hidden

  // one-time packing into WMMA-native bf16 layouts
  k_pack_A<<<2048, 256, 0, stream>>>(c, cA, 512, 32, H_DIM);
  k_pack_B<<<768, 256, 0, stream>>>(w_ih_h, wBihh, 192, 32, H_DIM);
  k_pack_B<<<768, 256, 0, stream>>>(w_hh_h, wBhhh, 192, 32, H_DIM);
  k_pack_B<<<768, 256, 0, stream>>>(w_ih_p, wBihp, 192, 32, H_DIM);
  k_pack_B<<<768, 256, 0, stream>>>(w_hh_p, wBhhp, 192, 32, H_DIM);
  k_pack_B<<<264, 256, 0, stream>>>(p_w1, pw1B, 64, 33, H_DIM + A_DIM);
  k_pack_B<<<4, 256, 0, stream>>>(p_w2, pw2B, 1, 32, H_DIM);

  // parallel precompute of the non-recurrent GEMMs (~103 GFLOP)
  k_gi_gemm<<<3072, 256, 0, stream>>>(cA, wBihh, b_ih_h, gih);
  k_gi_gemm<<<3072, 256, 0, stream>>>(cA, wBihp, b_ih_p, gip);

  k_init<<<256, 256, 0, stream>>>(hxs, hbuf, polbuf);

  float* out_pdist  = (float*)d_out;
  float* out_policy = out_pdist + (size_t)T_DIM * N_DIM * A_DIM;
  float* out_hxs    = out_policy + (size_t)T_DIM * N_DIM * H_DIM;

  // serial scan: GRU-only per-step kernel (64 blocks x 256)
  const size_t SB = (size_t)N_DIM * H_DIM;
  for (int t = 0; t < T_DIM; ++t) {
    const float* h_in    = hbuf + (size_t)(t & 1) * SB;
    float*       h_out   = hbuf + (size_t)((t + 1) & 1) * SB;
    const float* pol_in  = polbuf + (size_t)(t & 1) * SB;
    float*       pol_out = polbuf + (size_t)((t + 1) & 1) * SB;
    k_step<<<64, 256, 0, stream>>>(gih, gip, wBhhh, wBhhp, b_hh_h, b_hh_p, masks,
                                   h_in, h_out, pol_in, pol_out, h_histP,
                                   out_policy, t);
  }

  // policy MLP for all T*N rows at once (off the critical path)
  k_mlp1<<<1024, 256, 0, stream>>>(h_histP, pw1B, p_b1, masks, prev_a, hiddenP);
  k_mlp2<<<64, 256, 0, stream>>>(hiddenP, pw2B, p_b2, out_pdist);

  k_epilogue<<<256, 256, 0, stream>>>(hbuf + (size_t)(T_DIM & 1) * SB,
                                      polbuf + (size_t)(T_DIM & 1) * SB, out_hxs);
}
